// MaskedSelfAttention_2164663517206
// MI455X (gfx1250) — compile-verified, matrix operand order checked by reading, not run on hardware
//
#include <hip/hip_runtime.h>
#include <hip/hip_bf16.h>

typedef __attribute__((ext_vector_type(16))) __bf16 v16bf;
typedef __attribute__((ext_vector_type(4)))  __bf16 v4bf;
typedef __attribute__((ext_vector_type(8)))  float  v8f;
typedef __attribute__((ext_vector_type(4)))  float  v4f;
typedef __attribute__((ext_vector_type(4)))  int    v4i;

#define B_SZ 2
#define L_SZ 2048
#define D_SZ 1024
#define H_SZ 16
#define DH_SZ 64
#define SCALING 0.125f

#define AS1 __attribute__((address_space(1)))
#define AS3 __attribute__((address_space(3)))

#if __has_builtin(__builtin_amdgcn_global_load_async_to_lds_b128)
#define HAVE_ASYNC_LDS 1
#else
#define HAVE_ASYNC_LDS 0
#endif

#if HAVE_ASYNC_LDS
#if __has_builtin(__builtin_amdgcn_s_wait_asynccnt)
#define WAIT_ASYNC(n) __builtin_amdgcn_s_wait_asynccnt(n)
#else
#define WAIT_ASYNC(n) asm volatile("s_wait_asynccnt %0" ::"n"(n) : "memory")
#endif
static __device__ __forceinline__ void async_ld128(const void* g, void* l) {
    __builtin_amdgcn_global_load_async_to_lds_b128((AS1 v4i*)g, (AS3 v4i*)l, 0, 0);
}
#endif

static __device__ __forceinline__ v8f wmma_bf16(v16bf a, v16bf b, v8f c) {
    return __builtin_amdgcn_wmma_f32_16x16x32_bf16(false, a, false, b, (short)0, c, false, false);
}

// ---- 16-lane butterfly reductions in pure VALU via DPP16 (no LDS traffic) ----
#if __has_builtin(__builtin_amdgcn_update_dpp)
template <int CTRL>
static __device__ __forceinline__ float dpp_xf(float v) {
    return __int_as_float(
        __builtin_amdgcn_update_dpp(0, __float_as_int(v), CTRL, 0xF, 0xF, true));
}
static __device__ __forceinline__ float red_max16(float x) {
    x = fmaxf(x, dpp_xf<0xB1>(x));   // quad_perm(1,0,3,2)  lane^1
    x = fmaxf(x, dpp_xf<0x4E>(x));   // quad_perm(2,3,0,1)  lane^2
    x = fmaxf(x, dpp_xf<0x141>(x));  // row_half_mirror     lane^4 (post-quad)
    x = fmaxf(x, dpp_xf<0x140>(x));  // row_mirror          lane^8 (post-half)
    return x;
}
static __device__ __forceinline__ float red_sum16(float x) {
    x += dpp_xf<0xB1>(x);
    x += dpp_xf<0x4E>(x);
    x += dpp_xf<0x141>(x);
    x += dpp_xf<0x140>(x);
    return x;
}
#else
static __device__ __forceinline__ float red_max16(float x) {
    for (int off = 1; off < 16; off <<= 1) x = fmaxf(x, __shfl_xor(x, off, 16));
    return x;
}
static __device__ __forceinline__ float red_sum16(float x) {
    for (int off = 1; off < 16; off <<= 1) x += __shfl_xor(x, off, 16);
    return x;
}
#endif

// ---------------------------------------------------------------------------
// Kernel 1: h_pe = hidden + pos_emb -> bf16 ; hidden -> bf16
// ---------------------------------------------------------------------------
__global__ __launch_bounds__(256) void prep_kernel(
    const float* __restrict__ hs, const float* __restrict__ pe,
    __bf16* __restrict__ hpe, __bf16* __restrict__ hid, int n4) {
    int i = blockIdx.x * blockDim.x + threadIdx.x;
    if (i >= n4) return;
    v4f h = *(const v4f*)(hs + i * 4);
    v4f p = *(const v4f*)(pe + i * 4);
    v4bf o0, o1;
#pragma unroll
    for (int j = 0; j < 4; ++j) {
        o0[j] = (__bf16)(h[j] + p[j]);
        o1[j] = (__bf16)(h[j]);
    }
    *(v4bf*)(hpe + i * 4) = o0;
    *(v4bf*)(hid + i * 4) = o1;
}

// ---------------------------------------------------------------------------
// Kernel 2: weight f32 -> bf16
// ---------------------------------------------------------------------------
__global__ __launch_bounds__(256) void convw_kernel(
    const float* __restrict__ w, __bf16* __restrict__ wb, int n4) {
    int i = blockIdx.x * blockDim.x + threadIdx.x;
    if (i >= n4) return;
    v4f x = *(const v4f*)(w + i * 4);
    v4bf o;
#pragma unroll
    for (int j = 0; j < 4; ++j) o[j] = (__bf16)(x[j]);
    *(v4bf*)(wb + i * 4) = o;
}

// ---------------------------------------------------------------------------
// Kernel 3: tiled GEMM  out = (X @ W^T + bias) * scale
//   Block tile 128(M) x 64(N), 8 waves, each wave a 32x32 micro-tile.
//   Double-buffered LDS fed by GLOBAL_LOAD_ASYNC_TO_LDS_B128 (ASYNCcnt).
//   MODE 0: bf16 (M,N)   MODE 1: bf16 per-head transposed   MODE 2: f32 (M,N)
// ---------------------------------------------------------------------------
template <int MODE>
__global__ __launch_bounds__(256) void gemm_bf16_kernel(
    const __bf16* __restrict__ X, const __bf16* __restrict__ W,
    const float* __restrict__ bias, void* __restrict__ out,
    int M, int N, int K, float scale) {

#if HAVE_ASYNC_LDS
    __shared__ alignas(64) __bf16 sX[2][128][32];
    __shared__ alignas(64) __bf16 sW[2][64][32];
#else
    __shared__ alignas(64) __bf16 sX[1][128][32];
    __shared__ alignas(64) __bf16 sW[1][64][32];
#endif

    const int t    = threadIdx.x;
    const int lane = t & 31;
    const int wave = t >> 5;
    const int wm   = wave & 3;   // M quadrant (rows wm*32)
    const int wn   = wave >> 2;  // N half     (cols wn*32)
    const int lm   = lane & 15;
    const int lh   = lane >> 4;

    const int rowbase = blockIdx.y * 128;
    const int colbase = blockIdx.x * 64;
    const int ksteps  = K >> 5;

    v8f acc[2][2];
#pragma unroll
    for (int i = 0; i < 2; ++i)
#pragma unroll
        for (int j = 0; j < 2; ++j) acc[i][j] = (v8f){};

#if HAVE_ASYNC_LDS
    const int xr = t >> 1, xh = t & 1;  // X: 128 rows x 32B halves (2 x b128)
    const int wr = t >> 2, wc = t & 3;  // W: 64 rows x 16B chunks  (1 x b128)
    auto issue_tile = [&](int kt, int buf) {
        const __bf16* gx = X + (size_t)(rowbase + xr) * K + kt * 32 + xh * 16;
        async_ld128(gx,     &sX[buf][xr][xh * 16]);
        async_ld128(gx + 8, &sX[buf][xr][xh * 16 + 8]);
        const __bf16* gw = W + (size_t)(colbase + wr) * K + kt * 32 + wc * 8;
        async_ld128(gw, &sW[buf][wr][wc * 8]);
    };
    issue_tile(0, 0);
#endif

    for (int kt = 0; kt < ksteps; ++kt) {
#if HAVE_ASYNC_LDS
        const int cur = kt & 1;
        if (kt + 1 < ksteps) {
            __syncthreads();              // all waves done reading buf cur^1
            issue_tile(kt + 1, cur ^ 1);  // stream next tile into it
            WAIT_ASYNC(3);                // tile kt (this wave's part) landed
        } else {
            WAIT_ASYNC(0);
        }
        __syncthreads();                  // tile kt fully in LDS (all waves)
#else
        const int cur = 0;
        {
            const int r  = t >> 1;
            const int hf = t & 1;
            *(v16bf*)(&sX[0][r][hf * 16]) =
                *(const v16bf*)(X + (size_t)(rowbase + r) * K + kt * 32 + hf * 16);
            if (t < 128)
                *(v16bf*)(&sW[0][r][hf * 16]) =
                    *(const v16bf*)(W + (size_t)(colbase + r) * K + kt * 32 + hf * 16);
        }
        __syncthreads();
#endif

        v16bf a[2], b[2];
#pragma unroll
        for (int i = 0; i < 2; ++i)
            a[i] = *(const v16bf*)(&sX[cur][wm * 32 + i * 16 + lm][lh * 16]);
#pragma unroll
        for (int j = 0; j < 2; ++j)
            b[j] = *(const v16bf*)(&sW[cur][wn * 32 + j * 16 + lm][lh * 16]);

#pragma unroll
        for (int i = 0; i < 2; ++i)
#pragma unroll
            for (int j = 0; j < 2; ++j) acc[i][j] = wmma_bf16(a[i], b[j], acc[i][j]);

#if !HAVE_ASYNC_LDS
        __syncthreads();
#endif
    }

    // Store. C/D layout: VGPR v, lanes 0-15 -> M=v, lanes 16-31 -> M=v+8; N=lane%16
#pragma unroll
    for (int i = 0; i < 2; ++i) {
#pragma unroll
        for (int j = 0; j < 2; ++j) {
            const int col = colbase + wn * 32 + j * 16 + lm;
            const float bv = bias[col];
#pragma unroll
            for (int v = 0; v < 8; ++v) {
                const int row = rowbase + wm * 32 + i * 16 + v + lh * 8;
                float val = (acc[i][j][v] + bv) * scale;
                if (MODE == 0) {
                    ((__bf16*)out)[(size_t)row * N + col] = (__bf16)val;
                } else if (MODE == 1) {
                    const int bb = row >> 11, l = row & 2047;
                    const int h = col >> 6, d = col & 63;
                    ((__bf16*)out)[((size_t)((bb * H_SZ + h) * DH_SZ + d) << 11) + l] = (__bf16)val;
                } else {
                    ((float*)out)[(size_t)row * N + col] = val;
                }
            }
        }
    }
}

// ---------------------------------------------------------------------------
// Kernel 4: flash attention. One wave per (b, h, 16-query block).
// ---------------------------------------------------------------------------
__global__ __launch_bounds__(256) void flash_kernel(
    const __bf16* __restrict__ Q, const __bf16* __restrict__ Kb,
    const __bf16* __restrict__ Vt, const float* __restrict__ mask,
    __bf16* __restrict__ attn) {

    __shared__ alignas(64) __bf16 sP[8][16 * 32];  // per-wave P tile re-layout

    const int lane = threadIdx.x & 31;
    const int wave = threadIdx.x >> 5;
    const int lm   = lane & 15;
    const int lh   = lane >> 4;

    const int gw = blockIdx.x * 8 + wave;  // 0..4095
    const int b  = gw >> 11;
    const int h  = (gw >> 7) & 15;
    const int qbase = (gw & 127) * 16;

    const __bf16* qrow = Q + ((size_t)(b * L_SZ + qbase + lm) * D_SZ + h * DH_SZ);
    v16bf aq[2];
#pragma unroll
    for (int j = 0; j < 2; ++j) aq[j] = *(const v16bf*)(qrow + j * 32 + lh * 16);

    float mrow[8], lrow[8];
    v8f o[4];
#pragma unroll
    for (int v = 0; v < 8; ++v) { mrow[v] = -1e30f; lrow[v] = 0.f; }
#pragma unroll
    for (int d = 0; d < 4; ++d) o[d] = (v8f){};

    for (int kb = 0; kb < (L_SZ / 32); ++kb) {
        // ---- S = Q @ K^T for 32 keys (two 16-key N tiles) ----
        v8f s[2];
#pragma unroll
        for (int nt = 0; nt < 2; ++nt) {
            const __bf16* krow =
                Kb + ((size_t)(b * L_SZ + kb * 32 + nt * 16 + lm) * D_SZ + h * DH_SZ);
            v8f acc2 = (v8f){};
#pragma unroll
            for (int j = 0; j < 2; ++j) {
                v16bf bk = *(const v16bf*)(krow + j * 32 + lh * 16);
                acc2 = wmma_bf16(aq[j], bk, acc2);
            }
            s[nt] = acc2;
        }

        // ---- mask add ----
#pragma unroll
        for (int nt = 0; nt < 2; ++nt)
#pragma unroll
            for (int v = 0; v < 8; ++v) {
                const int srow = qbase + v + lh * 8;
                const int scol = kb * 32 + nt * 16 + lm;
                s[nt][v] += mask[(size_t)srow * L_SZ + scol];
            }

        // ---- online softmax: DPP16 butterfly reductions (pure VALU) ----
#pragma unroll
        for (int v = 0; v < 8; ++v) {
            const float mx   = red_max16(fmaxf(s[0][v], s[1][v]));
            const float mnew = fmaxf(mrow[v], mx);
            const float corr = __expf(mrow[v] - mnew);
            const float p0   = __expf(s[0][v] - mnew);
            const float p1   = __expf(s[1][v] - mnew);
            const float rs   = red_sum16(p0 + p1);
            lrow[v] = lrow[v] * corr + rs;
            mrow[v] = mnew;
#pragma unroll
            for (int d = 0; d < 4; ++d) o[d][v] *= corr;
            const int prow = v + lh * 8;  // D-layout -> LDS row-major 16x32
            sP[wave][prow * 32 + lm]      = (__bf16)p0;
            sP[wave][prow * 32 + 16 + lm] = (__bf16)p1;
        }
        __builtin_amdgcn_wave_barrier();  // LDS ops in-order within a wave

        // ---- O += P @ V (A from LDS; B from transposed V: contiguous) ----
        v16bf ap = *(const v16bf*)(&sP[wave][lm * 32 + lh * 16]);
#pragma unroll
        for (int d = 0; d < 4; ++d) {
            const __bf16* vrow =
                Vt + (((size_t)((b * H_SZ + h) * DH_SZ + d * 16 + lm)) << 11) + kb * 32 + lh * 16;
            v16bf bv = *(const v16bf*)(vrow);
            o[d] = wmma_bf16(ap, bv, o[d]);
        }
        __builtin_amdgcn_wave_barrier();
    }

    // ---- normalize and store (merge heads: attn[b][l][h*64+d]) ----
#pragma unroll
    for (int v = 0; v < 8; ++v) {
        const float inv = 1.0f / lrow[v];
        const int row = qbase + v + lh * 8;
#pragma unroll
        for (int d = 0; d < 4; ++d) {
            const int col = h * DH_SZ + d * 16 + lm;
            attn[(size_t)(b * L_SZ + row) * D_SZ + col] = (__bf16)(o[d][v] * inv);
        }
    }
}

// ---------------------------------------------------------------------------
// Host launcher
// ---------------------------------------------------------------------------
extern "C" void kernel_launch(void* const* d_in, const int* in_sizes, int n_in,
                              void* d_out, int out_size, void* d_ws, size_t ws_size,
                              hipStream_t stream) {
    (void)in_sizes; (void)n_in; (void)out_size; (void)ws_size;

    const float* hs   = (const float*)d_in[0];
    const float* pe   = (const float*)d_in[1];
    const float* mask = (const float*)d_in[2];
    const float* Wq   = (const float*)d_in[3];
    const float* bq   = (const float*)d_in[4];
    const float* Wk   = (const float*)d_in[5];
    const float* bk   = (const float*)d_in[6];
    const float* Wv   = (const float*)d_in[7];
    const float* bv   = (const float*)d_in[8];
    const float* Wo   = (const float*)d_in[9];
    const float* bo   = (const float*)d_in[10];
    float* out = (float*)d_out;

    const size_t MT  = (size_t)B_SZ * L_SZ;                   // 4096 rows
    const size_t ACT = MT * D_SZ * sizeof(__bf16);            // 8 MB
    const size_t WB  = (size_t)D_SZ * D_SZ * sizeof(__bf16);  // 2 MB

    char* ws = (char*)d_ws;
    __bf16* hpe  = (__bf16*)(ws); ws += ACT;
    __bf16* hid  = (__bf16*)(ws); ws += ACT;
    __bf16* wqb  = (__bf16*)(ws); ws += WB;
    __bf16* wkb  = (__bf16*)(ws); ws += WB;
    __bf16* wvb  = (__bf16*)(ws); ws += WB;
    __bf16* wob  = (__bf16*)(ws); ws += WB;
    __bf16* qbuf = (__bf16*)(ws); ws += ACT;
    __bf16* kbuf = (__bf16*)(ws); ws += ACT;
    __bf16* vtb  = (__bf16*)(ws); ws += ACT;
    __bf16* attn = (__bf16*)(ws); ws += ACT;

    {
        int n4 = (int)(MT * D_SZ / 4);
        prep_kernel<<<(n4 + 255) / 256, 256, 0, stream>>>(hs, pe, hpe, hid, n4);
    }
    {
        int n4 = D_SZ * D_SZ / 4;
        int g = (n4 + 255) / 256;
        convw_kernel<<<g, 256, 0, stream>>>(Wq, wqb, n4);
        convw_kernel<<<g, 256, 0, stream>>>(Wk, wkb, n4);
        convw_kernel<<<g, 256, 0, stream>>>(Wv, wvb, n4);
        convw_kernel<<<g, 256, 0, stream>>>(Wo, wob, n4);
    }
    dim3 gg(D_SZ / 64, (unsigned)(MT / 128));
    gemm_bf16_kernel<0><<<gg, 256, 0, stream>>>(hpe, wqb, bq, qbuf,
                                                (int)MT, D_SZ, D_SZ, SCALING);
    gemm_bf16_kernel<0><<<gg, 256, 0, stream>>>(hpe, wkb, bk, kbuf,
                                                (int)MT, D_SZ, D_SZ, 1.0f);
    gemm_bf16_kernel<1><<<gg, 256, 0, stream>>>(hid, wvb, bv, vtb,
                                                (int)MT, D_SZ, D_SZ, 1.0f);
    flash_kernel<<<512, 256, 0, stream>>>(qbuf, kbuf, vtb, mask, attn);
    gemm_bf16_kernel<2><<<gg, 256, 0, stream>>>(attn, wob, bo, (void*)out,
                                                (int)MT, D_SZ, D_SZ, 1.0f);
}
